// GRUAutoEncoder_53961969107150
// MI455X (gfx1250) — matrix operand order repaired and checked by reading, not compiled, not run on hardware
//
#include <hip/hip_runtime.h>
#include <hip/hip_bf16.h>
#include <stdint.h>

typedef __attribute__((ext_vector_type(16))) _Float16 v16h;
typedef __attribute__((ext_vector_type(8)))  float    v8f;
typedef __attribute__((ext_vector_type(4)))  float    v4f;

#define WMMA(a,b,c) __builtin_amdgcn_wmma_f32_16x16x32_f16(false,(a),false,(b),(short)0,(c),false,false)

constexpr int BSZ = 256, T = 1024, D = 64, H = 128, ICD = 100, FD = 3;
constexpr int HSTR = 132;  // halves per row for 16x128 LDS buffers (bank-skewed)
constexpr int XSTR = 68;   // halves per row for 16x64 x buffer

// Fast transcendentals: v_exp_f32 + v_rcp_f32, no IEEE divide expansion.
__device__ __forceinline__ float frcp_(float x) { return __builtin_amdgcn_rcpf(x); }
__device__ __forceinline__ float sigm_(float x) { return frcp_(1.f + __expf(-x)); }
__device__ __forceinline__ float tanh_(float x) { return 2.f * frcp_(1.f + __expf(-2.f * x)) - 1.f; }

__device__ __forceinline__ v8f bcast8(float s) {
  v8f v;
  #pragma unroll
  for (int i = 0; i < 8; ++i) v[i] = s;
  return v;
}

// Build a 16x32 f16 A-fragment from a row-major f16 LDS buffer (row = M, stride halves).
// ISA layout: lane M=L%16, half=L/16; VGPR i<4: K=2i,2i+1 (+8*half); i>=4: K=16+2(i-4) (+8*half).
__device__ __forceinline__ v16h ldsA(const _Float16* base, int stride, int k0, int m, int hf) {
  union { v16h v; uint32_t u[8]; } a;
  #pragma unroll
  for (int i = 0; i < 8; ++i) {
    int k = k0 + ((i >> 2) << 4) + (hf << 3) + ((i & 3) << 1);
    a.u[i] = *(const uint32_t*)(base + m * stride + k);
  }
  return a.v;
}

// Build a 32x16 f16 B-fragment from row-major f32 global W[R][ldw]; column N of the tile = W row (row0+n).
// ISA layout (matches A): half=L/16; element i<8: K=k0+8*half+i; i>=8: K=k0+16+8*half+(i-8).
__device__ __forceinline__ v16h gloB(const float* W, int ldw, int row0, int k0,
                                     int nvalid, int kmax, int n, int hf) {
  union { v16h v; _Float16 h[16]; } b;
  #pragma unroll
  for (int i = 0; i < 16; ++i) {
    int k = k0 + ((i >> 3) << 4) + (hf << 3) + (i & 7);
    float val = 0.f;
    if (n < nvalid && k < kmax) val = W[(size_t)(row0 + n) * ldw + k];
    b.h[i] = (_Float16)val;
  }
  return b.v;
}

__global__ __launch_bounds__(256, 1)
void lfads_fused_kernel(const float* __restrict__ mu,
                        const float* __restrict__ We_ih, const float* __restrict__ We_hh,
                        const float* __restrict__ be_ih, const float* __restrict__ be_hh,
                        const float* __restrict__ W_ic,  const float* __restrict__ b_ic,
                        const float* __restrict__ Wg_ih, const float* __restrict__ Wg_hh,
                        const float* __restrict__ bg_ih, const float* __restrict__ bg_hh,
                        const float* __restrict__ W_gf,  const float* __restrict__ b_gf,
                        const float* __restrict__ W_fr,  const float* __restrict__ b_fr,
                        float* __restrict__ out) {
  __shared__ _Float16 lds_h[16 * HSTR];   // current hidden state, f16, row-major
  __shared__ _Float16 lds_x[16 * XSTR];   // staged x_t, f16
  __shared__ _Float16 lds_ic[16 * HSTR];  // IC padded to 16x128, f16
  __shared__ float    lds_fact[16 * 4];   // exp(factors) per row

  const int tid  = threadIdx.x;
  const int lane = tid & 31;
  const int w    = tid >> 5;        // wave id 0..7; owns hidden cols [16w,16w+16)
  const int n16  = lane & 15;
  const int hf   = lane >> 4;
  const int b0   = blockIdx.x * 16; // batch tile
  const int c    = w * 16 + n16;    // hidden column owned by this lane

  for (int i = tid; i < 16 * HSTR; i += 256) { lds_h[i] = (_Float16)0.f; lds_ic[i] = (_Float16)0.f; }

  // ---------------- encoder weights -> registers (per-wave gate rows) ----------------
  v16h Wih_r[2], Wih_z[2], Wih_n[2];
  #pragma unroll
  for (int kt = 0; kt < 2; ++kt) {
    Wih_r[kt] = gloB(We_ih, D,           w * 16, kt * 32, 16, D, n16, hf);
    Wih_z[kt] = gloB(We_ih, D, H       + w * 16, kt * 32, 16, D, n16, hf);
    Wih_n[kt] = gloB(We_ih, D, 2 * H   + w * 16, kt * 32, 16, D, n16, hf);
  }
  v16h Whh_r[4], Whh_z[4], Whh_n[4];
  #pragma unroll
  for (int kt = 0; kt < 4; ++kt) {
    Whh_r[kt] = gloB(We_hh, H,           w * 16, kt * 32, 16, H, n16, hf);
    Whh_z[kt] = gloB(We_hh, H, H       + w * 16, kt * 32, 16, H, n16, hf);
    Whh_n[kt] = gloB(We_hh, H, 2 * H   + w * 16, kt * 32, 16, H, n16, hf);
  }
  const float b_r  = be_ih[c]         + be_hh[c];
  const float b_z  = be_ih[H + c]     + be_hh[H + c];
  const float b_nx = be_ih[2 * H + c];
  const float b_nh = be_hh[2 * H + c];

  v8f h = bcast8(0.f);  // this wave's f32 hidden tile (D-layout: row = r + 8*hf, col = c)

  const int xrow = tid >> 4;         // staging: row 0..15
  const int xcol = (tid & 15) * 4;   // staging: 4 contiguous floats
  __syncthreads();

  // ================= encoder GRU: T dependent steps =================
  #pragma unroll 1
  for (int t = 0; t < T; ++t) {
    // stage x_t (coalesced float4 per thread), convert to f16 in LDS
    const v4f xv = *(const v4f*)(mu + ((size_t)(b0 + xrow) * T + t) * D + xcol);
    lds_x[xrow * XSTR + xcol + 0] = (_Float16)xv.x;
    lds_x[xrow * XSTR + xcol + 1] = (_Float16)xv.y;
    lds_x[xrow * XSTR + xcol + 2] = (_Float16)xv.z;
    lds_x[xrow * XSTR + xcol + 3] = (_Float16)xv.w;
    if (t + 1 < T)
      __builtin_prefetch(mu + ((size_t)(b0 + xrow) * T + (t + 1)) * D + xcol, 0, 0);
    __syncthreads();  // x_t + previous h visible

    v16h Ax[2], Ah[4];
    Ax[0] = ldsA(lds_x, XSTR, 0, n16, hf);
    Ax[1] = ldsA(lds_x, XSTR, 32, n16, hf);
    #pragma unroll
    for (int kt = 0; kt < 4; ++kt) Ah[kt] = ldsA(lds_h, HSTR, kt * 32, n16, hf);

    v8f ar = bcast8(b_r), az = bcast8(b_z), anx = bcast8(b_nx), anh = bcast8(b_nh);
    #pragma unroll
    for (int kt = 0; kt < 2; ++kt) {
      ar  = WMMA(Ax[kt], Wih_r[kt], ar);
      az  = WMMA(Ax[kt], Wih_z[kt], az);
      anx = WMMA(Ax[kt], Wih_n[kt], anx);
    }
    #pragma unroll
    for (int kt = 0; kt < 4; ++kt) {
      ar  = WMMA(Ah[kt], Whh_r[kt], ar);
      az  = WMMA(Ah[kt], Whh_z[kt], az);
      anh = WMMA(Ah[kt], Whh_n[kt], anh);
    }
    __syncthreads();  // all LDS reads of this step done before h is overwritten

    #pragma unroll
    for (int rr = 0; rr < 8; ++rr) {
      float r  = sigm_(ar[rr]);
      float z  = sigm_(az[rr]);
      float nn = tanh_(anx[rr] + r * anh[rr]);
      float hv = (1.f - z) * nn + z * h[rr];
      h[rr] = hv;
      lds_h[(rr + 8 * hf) * HSTR + c] = (_Float16)hv;
    }
  }
  __syncthreads();  // h_enc visible in lds_h

  // ================= IC layer: IC = h_enc @ W_ic.T + b_ic (padded to 16x128 in LDS) =================
  if (w < 7) {
    const int nv  = (ICD - w * 16) < 16 ? (ICD - w * 16) : 16;
    const int icc = w * 16 + n16;
    v8f acc = bcast8(icc < ICD ? b_ic[icc] : 0.f);
    #pragma unroll
    for (int kt = 0; kt < 4; ++kt) {
      v16h At = ldsA(lds_h, HSTR, kt * 32, n16, hf);
      v16h Bt = gloB(W_ic, H, w * 16, kt * 32, nv, H, n16, hf);
      acc = WMMA(At, Bt, acc);
    }
    #pragma unroll
    for (int rr = 0; rr < 8; ++rr)
      lds_ic[(rr + 8 * hf) * HSTR + (w * 16 + n16)] = (_Float16)acc[rr];
  }
  __syncthreads();

  // ================= ic_proj = IC @ Wg_ih.T + biases (constant generator input) =================
  v8f icp_r = bcast8(bg_ih[c] + bg_hh[c]);
  v8f icp_z = bcast8(bg_ih[H + c] + bg_hh[H + c]);
  v8f icp_n = bcast8(bg_ih[2 * H + c]);
  const float g_bnh = bg_hh[2 * H + c];
  #pragma unroll
  for (int kt = 0; kt < 4; ++kt) {
    v16h At = ldsA(lds_ic, HSTR, kt * 32, n16, hf);
    icp_r = WMMA(At, gloB(Wg_ih, ICD,           w * 16, kt * 32, 16, ICD, n16, hf), icp_r);
    icp_z = WMMA(At, gloB(Wg_ih, ICD, H       + w * 16, kt * 32, 16, ICD, n16, hf), icp_z);
    icp_n = WMMA(At, gloB(Wg_ih, ICD, 2 * H   + w * 16, kt * 32, 16, ICD, n16, hf), icp_n);
  }

  // generator recurrent weights replace encoder ones in the same registers
  #pragma unroll
  for (int kt = 0; kt < 4; ++kt) {
    Whh_r[kt] = gloB(Wg_hh, H,           w * 16, kt * 32, 16, H, n16, hf);
    Whh_z[kt] = gloB(Wg_hh, H, H       + w * 16, kt * 32, 16, H, n16, hf);
    Whh_n[kt] = gloB(Wg_hh, H, 2 * H   + w * 16, kt * 32, 16, H, n16, hf);
  }

  // wave 0: factors readout weights (W_gf zero-padded 3 -> 16 rows)
  v16h Wgf[4];
  float bgf = 0.f;
  if (w == 0) {
    #pragma unroll
    for (int kt = 0; kt < 4; ++kt) Wgf[kt] = gloB(W_gf, H, 0, kt * 32, FD, H, n16, hf);
    if (n16 < FD) bgf = b_gf[n16];
  }

  // rates readout constants per thread (4 output columns each)
  const int rrow = tid >> 4;
  const int rc0  = (tid & 15) * 4;
  float wfr[4][3], bfr4[4];
  #pragma unroll
  for (int j = 0; j < 4; ++j) {
    #pragma unroll
    for (int f = 0; f < FD; ++f) wfr[j][f] = W_fr[(rc0 + j) * FD + f];
    bfr4[j] = b_fr[rc0 + j];
  }
  __syncthreads();

  // ================= generator GRU + fused readout: T dependent steps =================
  #pragma unroll 1
  for (int t = 0; t < T; ++t) {
    v16h Ah[4];
    #pragma unroll
    for (int kt = 0; kt < 4; ++kt) Ah[kt] = ldsA(lds_h, HSTR, kt * 32, n16, hf);

    v8f ar = icp_r, az = icp_z, anh = bcast8(g_bnh);
    #pragma unroll
    for (int kt = 0; kt < 4; ++kt) {
      ar  = WMMA(Ah[kt], Whh_r[kt], ar);
      az  = WMMA(Ah[kt], Whh_z[kt], az);
      anh = WMMA(Ah[kt], Whh_n[kt], anh);
    }
    __syncthreads();  // lds_h reads done

    #pragma unroll
    for (int rr = 0; rr < 8; ++rr) {
      float r  = sigm_(ar[rr]);
      float z  = sigm_(az[rr]);
      float nn = tanh_(icp_n[rr] + r * anh[rr]);
      float hv = (1.f - z) * nn + z * h[rr];
      h[rr] = hv;
      lds_h[(rr + 8 * hf) * HSTR + c] = (_Float16)hv;
    }
    __syncthreads();  // new h visible

    if (w == 0) {  // factors = h @ W_gf.T + b_gf ; exp ; to LDS
      v8f f = bcast8(bgf);
      #pragma unroll
      for (int kt = 0; kt < 4; ++kt)
        f = WMMA(ldsA(lds_h, HSTR, kt * 32, n16, hf), Wgf[kt], f);
      #pragma unroll
      for (int rr = 0; rr < 8; ++rr) {
        float e = __expf(f[rr]);
        if (n16 < FD) lds_fact[(rr + 8 * hf) * 4 + n16] = e;
      }
    }
    __syncthreads();  // factors visible

    // rates[b, t, :] = exp(factors) @ W_fr.T + b_fr  (streaming NT store)
    const float e0 = lds_fact[rrow * 4 + 0];
    const float e1 = lds_fact[rrow * 4 + 1];
    const float e2 = lds_fact[rrow * 4 + 2];
    v4f o;
    o.x = e0 * wfr[0][0] + e1 * wfr[0][1] + e2 * wfr[0][2] + bfr4[0];
    o.y = e0 * wfr[1][0] + e1 * wfr[1][1] + e2 * wfr[1][2] + bfr4[1];
    o.z = e0 * wfr[2][0] + e1 * wfr[2][1] + e2 * wfr[2][2] + bfr4[2];
    o.w = e0 * wfr[3][0] + e1 * wfr[3][1] + e2 * wfr[3][2] + bfr4[3];
    __builtin_nontemporal_store(o, (v4f*)(out + ((size_t)(b0 + rrow) * T + t) * D + rc0));
  }
}

extern "C" void kernel_launch(void* const* d_in, const int* in_sizes, int n_in,
                              void* d_out, int out_size, void* d_ws, size_t ws_size,
                              hipStream_t stream) {
  (void)in_sizes; (void)n_in; (void)out_size; (void)d_ws; (void)ws_size;
  const float* mu    = (const float*)d_in[0];
  const float* We_ih = (const float*)d_in[1];
  const float* We_hh = (const float*)d_in[2];
  const float* be_ih = (const float*)d_in[3];
  const float* be_hh = (const float*)d_in[4];
  const float* W_ic  = (const float*)d_in[5];
  const float* b_ic  = (const float*)d_in[6];
  const float* Wg_ih = (const float*)d_in[7];
  const float* Wg_hh = (const float*)d_in[8];
  const float* bg_ih = (const float*)d_in[9];
  const float* bg_hh = (const float*)d_in[10];
  const float* W_gf  = (const float*)d_in[11];
  const float* b_gf  = (const float*)d_in[12];
  const float* W_fr  = (const float*)d_in[13];
  const float* b_fr  = (const float*)d_in[14];
  float* out = (float*)d_out;

  lfads_fused_kernel<<<BSZ / 16, 256, 0, stream>>>(
      mu, We_ih, We_hh, be_ih, be_hh, W_ic, b_ic,
      Wg_ih, Wg_hh, bg_ih, bg_hh, W_gf, b_gf, W_fr, b_fr, out);
}
